// DeformableConv2d_76321568850098
// MI455X (gfx1250) — compile-verified, matrix operand order checked by reading, not run on hardware
//
#include <hip/hip_runtime.h>
#include <math.h>

typedef _Float16 v16h __attribute__((ext_vector_type(16)));
typedef _Float16 v8h  __attribute__((ext_vector_type(8)));
typedef _Float16 v4h  __attribute__((ext_vector_type(4)));
typedef float    v8f  __attribute__((ext_vector_type(8)));

#define Bn   4
#define Cc   256
#define HW   4096      // 64*64
#define CK   2304      // Cc*9
#define Oo   256
#define NCH  72        // CK/32

// workspace layout (bytes); total ~3.1 MB
#define WS_W16   0
#define WS_W27   (WS_W16 + Oo*CK*2)          // main weight f16: 1,179,648 B
#define WS_OFF   (WS_W27 + 32*CK*2)          // 27(+pad) rows f16: 147,456 B
#define WS_MASK  (WS_OFF + Bn*18*HW*4)       // offset f32: 1,179,648 B
// mask f32: 589,824 B -> end 3,096,576 B

__device__ __forceinline__ v16h cat16(v8h lo, v8h hi) {
  return __builtin_shufflevector(lo, hi, 0,1,2,3,4,5,6,7,8,9,10,11,12,13,14,15);
}
__device__ __forceinline__ v8f wmma16(v16h a, v16h b, v8f c) {
  return __builtin_amdgcn_wmma_f32_16x16x32_f16(false, a, false, b, (short)0, c, false, false);
}
// exact kk/3 for kk in [0,8]
__device__ __forceinline__ int div3_9(int kk) { return (kk * 11) >> 5; }
// fast sigmoid: v_exp_f32 + v_rcp_f32 (mask feeds interpolation; 1-ulp is plenty)
__device__ __forceinline__ float fast_sigmoid(float t) {
  return __builtin_amdgcn_rcpf(1.f + __expf(-t));
}

// ---------------------------------------------------------------- converts
__global__ void k_convert(const float* __restrict__ w, const float* __restrict__ offw,
                          const float* __restrict__ maskw,
                          _Float16* __restrict__ w16, _Float16* __restrict__ w27) {
  int i = blockIdx.x * blockDim.x + threadIdx.x;
  const int N1 = Oo * CK, N2 = 32 * CK;
  for (; i < N1 + N2; i += gridDim.x * blockDim.x) {
    if (i < N1) {
      w16[i] = (_Float16)w[i];
    } else {
      int j = i - N1; int row = j / CK; int ck = j - row * CK;
      float v = 0.f;
      if (row < 18)      v = offw[row * CK + ck];
      else if (row < 27) v = maskw[(row - 18) * CK + ck];
      w27[j] = (_Float16)v;
    }
  }
}

// ------------------------------------------------- offset/mask conv (WMMA)
__global__ __launch_bounds__(256)
void k_offmask(const float* __restrict__ x, const _Float16* __restrict__ w27,
               const float* __restrict__ offb, const float* __restrict__ maskb,
               float* __restrict__ offo, float* __restrict__ masko) {
  __shared__ _Float16 xch[2][128][40];       // double-buffered, padded stride
  const int b     = blockIdx.x >> 5;         // 32 tiles of 128 positions
  const int pBase = (blockIdx.x & 31) * 128;
  const int tid = threadIdx.x, lane = tid & 31, wv = tid >> 5;
  const float* xb = x + (size_t)b * Cc * HW;

  v8f acc0 = {}; v8f acc1 = {};
  const int p_l = tid >> 1;
  const int kh  = (tid & 1) * 16;
  const int p   = pBase + p_l;
  const int y = p >> 6, xo = p & 63;
  const int rsel = (lane >> 4) * 8;
  const int rlo  = lane & 15;

  // chunk-start (c,kk) for K = kc*32 + kh, updated incrementally (32 = 3*9 + 5)
  int cS = (kh == 0) ? 0 : 1;
  int kkS = (kh == 0) ? 0 : 7;

  _Float16 hv[16];
  // Per-element indices are derived INDEPENDENTLY from chunk-start state:
  // kkj = kkS + j <= 23 needs at most two 9-wraps -> two parallel compares,
  // no cross-element dependency chain.
#define OM_GATHER()                                                            \
  {                                                                            \
    _Pragma("unroll")                                                          \
    for (int j = 0; j < 16; ++j) {                                             \
      int kkj = kkS + j;                                                       \
      int w1 = (kkj >= 9) ? 1 : 0, w2 = (kkj >= 18) ? 1 : 0;                   \
      int c  = cS + w1 + w2;                                                   \
      int kkm = kkj - 9 * (w1 + w2);                                           \
      int ky = div3_9(kkm), kx = kkm - ky * 3;                                 \
      int sy = y + ky - 1, sx = xo + kx - 1;                                   \
      bool ok = ((unsigned)sy < 64u) & ((unsigned)sx < 64u);                   \
      int syc = min(max(sy, 0), 63), sxc = min(max(sx, 0), 63);                \
      float v = xb[(c << 12) + (syc << 6) + sxc];                              \
      hv[j] = ok ? (_Float16)v : (_Float16)0.f;                                \
    }                                                                          \
    kkS += 5; cS += 3; if (kkS >= 9) { kkS -= 9; cS += 1; }                    \
  }

  // prologue: chunk 0
  OM_GATHER();
  *(v8h*)&xch[0][p_l][kh]     = *(v8h*)&hv[0];
  *(v8h*)&xch[0][p_l][kh + 8] = *(v8h*)&hv[8];
  __syncthreads();

  for (int kc = 0; kc < NCH - 1; ++kc) {
    const int ck0 = kc * 32;
    const int cur = kc & 1;
    OM_GATHER();                               // next chunk's loads in flight
    // A fragments (rows 0..31 of the 27-row weight), B from LDS chunk
    const _Float16* wp0 = w27 + (size_t)(rlo)      * CK + ck0 + rsel;
    const _Float16* wp1 = w27 + (size_t)(16 + rlo) * CK + ck0 + rsel;
    v16h a0 = cat16(*(const v8h*)wp0, *(const v8h*)(wp0 + 16));
    v16h a1 = cat16(*(const v8h*)wp1, *(const v8h*)(wp1 + 16));
    int brow = wv * 16 + rlo;
    v16h bf = cat16(*(v8h*)&xch[cur][brow][rsel], *(v8h*)&xch[cur][brow][rsel + 16]);
    acc0 = wmma16(a0, bf, acc0);
    acc1 = wmma16(a1, bf, acc1);
    *(v8h*)&xch[cur ^ 1][p_l][kh]     = *(v8h*)&hv[0];
    *(v8h*)&xch[cur ^ 1][p_l][kh + 8] = *(v8h*)&hv[8];
    __syncthreads();
  }
  { // final chunk (buffer (NCH-1)&1 = 1)
    const int ck0 = (NCH - 1) * 32;
    const _Float16* wp0 = w27 + (size_t)(rlo)      * CK + ck0 + rsel;
    const _Float16* wp1 = w27 + (size_t)(16 + rlo) * CK + ck0 + rsel;
    v16h a0 = cat16(*(const v8h*)wp0, *(const v8h*)(wp0 + 16));
    v16h a1 = cat16(*(const v8h*)wp1, *(const v8h*)(wp1 + 16));
    int brow = wv * 16 + rlo;
    v16h bf = cat16(*(v8h*)&xch[1][brow][rsel], *(v8h*)&xch[1][brow][rsel + 16]);
    acc0 = wmma16(a0, bf, acc0);
    acc1 = wmma16(a1, bf, acc1);
  }
#undef OM_GATHER

  const int rhi = (lane >> 4) * 8;
  const int pp = pBase + wv * 16 + rlo;
  #pragma unroll
  for (int r = 0; r < 8; ++r) {
    int co0 = r + rhi;            // acc0 rows 0..15
    int co1 = 16 + r + rhi;       // acc1 rows 16..31
    float v0 = acc0[r], v1 = acc1[r];
    if (co0 < 18)       offo[((b * 18 + co0) << 12) + pp] = v0 + offb[co0];
    if (co1 < 18)       offo[((b * 18 + co1) << 12) + pp] = v1 + offb[co1];
    else if (co1 < 27)  masko[((b * 9 + co1 - 18) << 12) + pp] =
                          fast_sigmoid(v1 + maskb[co1 - 18]);
  }
}

// ------------------------------------- deformable sample + main GEMM (WMMA)
__global__ __launch_bounds__(256)
void k_main(const float* __restrict__ x, const _Float16* __restrict__ w16,
            const float* __restrict__ bias, const float* __restrict__ offo,
            const float* __restrict__ masko, float* __restrict__ out) {
  __shared__ float        mW[288][4];     // recombined 2x2-tap weights (mask+validity folded)
  __shared__ unsigned int mI[288];        // packed u16 row-pair base indices (top|bot<<16)
  __shared__ _Float16     samp[2][32][40];// double-buffered [pos][K] chunk, padded stride
  const int b     = blockIdx.x >> 7;      // 128 tiles of 32 positions
  const int pBase = (blockIdx.x & 127) * 32;
  const int tid = threadIdx.x, lane = tid & 31, wv = tid >> 5;

  // -------- per-(tap,position) metadata, computed once per block
  // Each bilinear row reads 2 adjacent pixels starting at xbase=clamp(x0,0,62).
  // A corner contributes iff it lands on xbase or xbase+1 (this encodes x-validity
  // exactly), so x-weights recombine onto the loaded pair with no approximation.
  for (int e = tid; e < 288; e += 256) {
    int kk = e >> 5, p_l = e & 31;
    int p = pBase + p_l;
    int y = p >> 6, xo = p & 63;
    int ky = div3_9(kk), kx = kk - ky * 3;
    float dy = offo[((b * 18 + 2 * kk)     << 12) + p];
    float dx = offo[((b * 18 + 2 * kk + 1) << 12) + p];
    float m  = masko[((b * 9 + kk)         << 12) + p];
    float py = (float)(y + ky - 1) + dy;
    float px = (float)(xo + kx - 1) + dx;
    float y0f = floorf(py), x0f = floorf(px);
    float fy = py - y0f, fx = px - x0f;
    int y0 = (int)y0f, x0 = (int)x0f, y1 = y0 + 1, x1 = x0 + 1;
    float wx0 = 1.f - fx, wx1 = fx;
    int xbase = min(max(x0, 0), 62);
    float wxA = 0.f, wxB = 0.f;
    if (x0 == xbase)          wxA += wx0;
    else if (x0 == xbase + 1) wxB += wx0;
    if (x1 == xbase)          wxA += wx1;
    else if (x1 == xbase + 1) wxB += wx1;
    float wT = m * (1.f - fy) * (((unsigned)y0 < 64u) ? 1.f : 0.f);
    float wB = m * fy         * (((unsigned)y1 < 64u) ? 1.f : 0.f);
    int cy0 = min(max(y0, 0), 63), cy1 = min(max(y1, 0), 63);
    mW[e][0] = wT * wxA;
    mW[e][1] = wT * wxB;
    mW[e][2] = wB * wxA;
    mW[e][3] = wB * wxB;
    mI[e] = (unsigned)((cy0 << 6) + xbase) | ((unsigned)((cy1 << 6) + xbase) << 16);
  }
  __syncthreads();

  v8f acc00 = {}, acc01 = {}, acc10 = {}, acc11 = {};
  const float* xb = x + (size_t)b * Cc * HW;
  const int p_l = tid >> 3;         // 4 consecutive K at one position
  const int k_l = (tid & 7) * 4;
  const int mBase = wv * 32;        // each wave owns 32 output channels
  const int rsel = (lane >> 4) * 8;
  const int rlo  = lane & 15;

  // chunk-start (c, kk, e) for K = kc*32 + k_l, updated incrementally
  int cS = k_l / 9;                 // once, outside the hot loop
  int kkS = k_l - cS * 9;
  int eS = (kkS << 5) + p_l;

  _Float16 hv[4];
  // kkj = kkS + j <= 11 -> at most one 9-wrap; each element independent.
#define MN_GATHER()                                                            \
  {                                                                            \
    _Pragma("unroll")                                                          \
    for (int j = 0; j < 4; ++j) {                                              \
      int kkj = kkS + j;                                                       \
      bool wrap = (kkj >= 9);                                                  \
      int c = cS + (wrap ? 1 : 0);                                             \
      int e = eS + (j << 5) - (wrap ? 288 : 0);                                \
      const float* xc = xb + (c << 12);                                        \
      float4   wq = *(const float4*)&mW[e][0];                                 \
      unsigned ii = mI[e];                                                     \
      float2 t  = *(const float2*)(xc + (ii & 0xffffu));                       \
      float2 bo = *(const float2*)(xc + (ii >> 16));                           \
      float v = wq.x * t.x + wq.y * t.y + wq.z * bo.x + wq.w * bo.y;           \
      hv[j] = (_Float16)v;                                                     \
    }                                                                          \
    kkS += 5; eS += 160; cS += 3;                                              \
    if (kkS >= 9) { kkS -= 9; eS -= 288; cS += 1; }                            \
  }

  // prologue: chunk 0
  MN_GATHER();
  *(v4h*)&samp[0][p_l][k_l] = *(v4h*)&hv[0];
  __syncthreads();

  for (int kc = 0; kc < NCH - 1; ++kc) {
    const int ck0 = kc * 32;
    const int cur = kc & 1;
    MN_GATHER();                               // next chunk's loads in flight
    // -------- WMMA: 2 M-tiles x 2 N-tiles per wave
    const _Float16* wp0 = w16 + (size_t)(mBase + rlo)      * CK + ck0 + rsel;
    const _Float16* wp1 = w16 + (size_t)(mBase + 16 + rlo) * CK + ck0 + rsel;
    v16h a0 = cat16(*(const v8h*)wp0, *(const v8h*)(wp0 + 16));
    v16h a1 = cat16(*(const v8h*)wp1, *(const v8h*)(wp1 + 16));
    v16h b0 = cat16(*(v8h*)&samp[cur][rlo][rsel],      *(v8h*)&samp[cur][rlo][rsel + 16]);
    v16h b1 = cat16(*(v8h*)&samp[cur][16 + rlo][rsel], *(v8h*)&samp[cur][16 + rlo][rsel + 16]);
    acc00 = wmma16(a0, b0, acc00);
    acc01 = wmma16(a0, b1, acc01);
    acc10 = wmma16(a1, b0, acc10);
    acc11 = wmma16(a1, b1, acc11);
    *(v4h*)&samp[cur ^ 1][p_l][k_l] = *(v4h*)&hv[0];
    __syncthreads();
  }
  { // final chunk (buffer (NCH-1)&1 = 1)
    const int ck0 = (NCH - 1) * 32;
    const _Float16* wp0 = w16 + (size_t)(mBase + rlo)      * CK + ck0 + rsel;
    const _Float16* wp1 = w16 + (size_t)(mBase + 16 + rlo) * CK + ck0 + rsel;
    v16h a0 = cat16(*(const v8h*)wp0, *(const v8h*)(wp0 + 16));
    v16h a1 = cat16(*(const v8h*)wp1, *(const v8h*)(wp1 + 16));
    v16h b0 = cat16(*(v8h*)&samp[1][rlo][rsel],      *(v8h*)&samp[1][rlo][rsel + 16]);
    v16h b1 = cat16(*(v8h*)&samp[1][16 + rlo][rsel], *(v8h*)&samp[1][16 + rlo][rsel + 16]);
    acc00 = wmma16(a0, b0, acc00);
    acc01 = wmma16(a0, b1, acc01);
    acc10 = wmma16(a1, b0, acc10);
    acc11 = wmma16(a1, b1, acc11);
  }
#undef MN_GATHER

  // -------- epilogue: D layout = VGPR r -> M = r (lanes<16) / 8+r (lanes>=16)
  const int rhi = (lane >> 4) * 8;
  const int pp0 = pBase + rlo, pp1 = pBase + 16 + rlo;
  #pragma unroll
  for (int r = 0; r < 8; ++r) {
    int o0 = mBase + r + rhi, o1 = mBase + 16 + r + rhi;
    float b0v = bias[o0], b1v = bias[o1];
    out[((size_t)(b * Oo + o0) << 12) + pp0] = acc00[r] + b0v;
    out[((size_t)(b * Oo + o0) << 12) + pp1] = acc01[r] + b0v;
    out[((size_t)(b * Oo + o1) << 12) + pp0] = acc10[r] + b1v;
    out[((size_t)(b * Oo + o1) << 12) + pp1] = acc11[r] + b1v;
  }
}

extern "C" void kernel_launch(void* const* d_in, const int* in_sizes, int n_in,
                              void* d_out, int out_size, void* d_ws, size_t ws_size,
                              hipStream_t stream) {
  const float* x     = (const float*)d_in[0];
  const float* w     = (const float*)d_in[1];
  const float* bias  = (const float*)d_in[2];
  const float* offw  = (const float*)d_in[3];
  const float* offb  = (const float*)d_in[4];
  const float* maskw = (const float*)d_in[5];
  const float* maskb = (const float*)d_in[6];

  char* ws = (char*)d_ws;
  _Float16* w16  = (_Float16*)(ws + WS_W16);
  _Float16* w27  = (_Float16*)(ws + WS_W27);
  float*    offo = (float*)(ws + WS_OFF);
  float*    masko= (float*)(ws + WS_MASK);
  float*    out  = (float*)d_out;

  k_convert<<<512, 256, 0, stream>>>(w, offw, maskw, w16, w27);
  k_offmask<<<128, 256, 0, stream>>>(x, w27, offb, maskb, offo, masko);
  k_main   <<<512, 256, 0, stream>>>(x, w16, bias, offo, masko, out);
}
